// QuantDecoderLayer_38689065402899
// MI455X (gfx1250) — matrix operand order repaired and checked by reading, not compiled, not run on hardware
//
#include <hip/hip_runtime.h>
#include <hip/hip_bf16.h>

typedef __bf16 bf16;
typedef __attribute__((ext_vector_type(16))) __bf16 v16bf;
typedef __attribute__((ext_vector_type(8)))  __bf16 bf16x8;
typedef __attribute__((ext_vector_type(8)))  float  v8f;

constexpr int B_  = 2;
constexpr int S_  = 1024;
constexpr int H_  = 2048;
constexpr int NH_ = 16;
constexpr int HD_ = 128;
constexpr int E_  = 16;
constexpr int TK_ = 4;
constexpr int I_  = 1408;
constexpr int I2_ = 5632;
constexpr int T_  = B_ * S_;
constexpr int SLOTS_ = T_ * TK_;  // 8192 (token,expert) slots

// ---------------- CDNA5 async global->LDS copy (ASYNCcnt path) ----------------
// Pure bf16 tile copies bypass the register file via the async copy engine.
// LDS operand is the low 32 bits of the flat shared pointer (= LDS byte offset).

__device__ __forceinline__ void async_copy_b128(void* lds_dst, const void* gsrc) {
  unsigned loff = (unsigned)(size_t)lds_dst;
  unsigned long long ga = (unsigned long long)(size_t)gsrc;
  asm volatile("global_load_async_to_lds_b128 %0, %1, off"
               :
               : "v"(loff), "v"(ga)
               : "memory");
}

__device__ __forceinline__ void wait_async() {
  asm volatile("s_wait_asynccnt 0x0" ::: "memory");
}

// ---------------- wave32 WMMA helpers (CDNA5 fragment layouts) ----------------

__device__ __forceinline__ v8f v8f_zero() {
  v8f z;
#pragma unroll
  for (int i = 0; i < 8; ++i) z[i] = 0.0f;
  return z;
}

__device__ __forceinline__ bf16x8 bf8_zero() {
  bf16x8 z;
#pragma unroll
  for (int i = 0; i < 8; ++i) z[i] = (bf16)0.0f;
  return z;
}

__device__ __forceinline__ v16bf combine16(bf16x8 lo, bf16x8 hi) {
  v16bf r;
#pragma unroll
  for (int i = 0; i < 8; ++i) { r[i] = lo[i]; r[i + 8] = hi[i]; }
  return r;
}

// A fragment 16x32 bf16: lane m (both halves) = row row0+m.
// half 0: elems 0..15 = K 0..15
// half 1: elems 0..15 = K 16..31
__device__ __forceinline__ v16bf load_a_frag(const bf16* base, int row0, int stride, int kbase) {
  const int lane = threadIdx.x & 31;
  const int hlf  = lane >> 4;
  const int m    = lane & 15;
  const bf16* p = base + (size_t)(row0 + m) * stride + kbase;
  bf16x8 lo = *(const bf16x8*)(p + hlf * 16);
  bf16x8 hi = *(const bf16x8*)(p + hlf * 16 + 8);
  return combine16(lo, hi);
}

// B fragment 32x16 bf16 from N-major storage Bt[n][k] (i.e. W[N,K] row-major):
// lane n = column row0+n; elems 0..15 = K kbase + hlf*16 .. +15 (contiguous).
__device__ __forceinline__ v16bf load_b_frag(const bf16* base, int row0, int stride, int kbase) {
  const int lane = threadIdx.x & 31;
  const int hlf  = lane >> 4;
  const int n    = lane & 15;
  const bf16* p = base + (size_t)(row0 + n) * stride + kbase + hlf * 16;
  return combine16(*(const bf16x8*)(p), *(const bf16x8*)(p + 8));
}

__device__ __forceinline__ v8f wmma_bf16(v16bf a, v16bf b, v8f c) {
  return __builtin_amdgcn_wmma_f32_16x16x32_bf16(false, a, false, b, (short)0, c, false, false);
}

__device__ __forceinline__ bf16x8 cvt8(const float* src) {
  bf16x8 r;
#pragma unroll
  for (int i = 0; i < 8; ++i) r[i] = (bf16)src[i];
  return r;
}

// ---------------- elementwise / reduction kernels ----------------

__global__ void zero_k(float* __restrict__ moe, int* __restrict__ cnt) {
  const size_t idx = (size_t)blockIdx.x * 256 + threadIdx.x;
  if (idx < (size_t)T_ * H_) moe[idx] = 0.0f;
  if (idx < E_) cnt[idx] = 0;
}

__global__ void rmsnorm_k(const float* __restrict__ X, const float* __restrict__ Wn,
                          bf16* __restrict__ Out) {
  const int t = blockIdx.x;
  const int tid = threadIdx.x;
  __shared__ float red[256];
  float s = 0.0f;
  for (int k = tid; k < H_; k += 256) {
    const float v = X[(size_t)t * H_ + k];
    s += v * v;
  }
  red[tid] = s;
  __syncthreads();
  for (int off = 128; off > 0; off >>= 1) {
    if (tid < off) red[tid] += red[tid + off];
    __syncthreads();
  }
  const float r = rsqrtf(red[0] / (float)H_ + 1e-6f);
  for (int k = tid; k < H_; k += 256)
    Out[(size_t)t * H_ + k] = (bf16)(X[(size_t)t * H_ + k] * r * Wn[k]);
}

// rotate-half RoPE; fp32 q/k [B,S,NH,HD] -> bf16 [B,NH,S,HD]
__global__ void rope_k(const float* __restrict__ Qf, const float* __restrict__ Kf,
                       const int* __restrict__ pos_ids, bf16* __restrict__ Qb,
                       bf16* __restrict__ Kb) {
  const int idx = blockIdx.x * 256 + threadIdx.x;  // [b][s][h][d], d<64
  const int d = idx & 63;
  const int h = (idx >> 6) & 15;
  const int s = (idx >> 10) & 1023;
  const int b = idx >> 20;
  if (b >= B_) return;
  const float p   = (float)pos_ids[b * S_ + s];
  const float inv = __expf(-(float)(2 * d) * (13.815510557964274f / 128.0f));  // theta=1e6
  const float fr  = p * inv;
  const float c = cosf(fr), sn = sinf(fr);
  const size_t ib = (((size_t)b * S_ + s) * NH_ + h) * HD_ + d;
  const size_t ob = (((size_t)b * NH_ + h) * S_ + s) * HD_ + d;
  const float q1 = Qf[ib], q2 = Qf[ib + 64];
  const float k1 = Kf[ib], k2 = Kf[ib + 64];
  Qb[ob]      = (bf16)(q1 * c - q2 * sn);
  Qb[ob + 64] = (bf16)(q2 * c + q1 * sn);
  Kb[ob]      = (bf16)(k1 * c - k2 * sn);
  Kb[ob + 64] = (bf16)(k2 * c + k1 * sn);
}

// V fp32 [B,S,NH,HD] -> bf16 [B*NH, HD, S] (so P@V B-frags are contiguous in keys)
__global__ void vtr_k(const float* __restrict__ Vf, bf16* __restrict__ Vt) {
  const int idx = blockIdx.x * 256 + threadIdx.x;  // [bh][d][s]
  const int s  = idx & (S_ - 1);
  const int d  = (idx >> 10) & (HD_ - 1);
  const int bh = idx >> 17;
  if (bh >= B_ * NH_) return;
  const int b = bh >> 4, h = bh & 15;
  Vt[idx] = (bf16)Vf[(((size_t)b * S_ + s) * NH_ + h) * HD_ + d];
}

// ---------------- dense GEMM: C[M,N] = A[M,K]bf16 @ W[N,K]^T (+res) ----------------
// block tile 128x64, 8 waves, each wave a 32x32 quad of 16x16 WMMA tiles
// A tile staged via async global->LDS copies; W tile converted fp32->bf16 in VGPRs.

__global__ __launch_bounds__(256) void gemm_f32(
    const bf16* __restrict__ A, const float* __restrict__ W, float* __restrict__ C,
    const float* __restrict__ res, int Kd, int N) {
  __shared__ bf16 As[128 * 40];
  __shared__ bf16 Bs[64 * 40];
  const int tid = threadIdx.x;
  const int w = tid >> 5, lane = tid & 31, hlf = lane >> 4, nn = lane & 15;
  const int waveM = w & 3, waveN = w >> 2;
  const int m0 = blockIdx.x * 128, n0 = blockIdx.y * 64;
  v8f acc[2][2];
#pragma unroll
  for (int i = 0; i < 2; ++i)
#pragma unroll
    for (int j = 0; j < 2; ++j) acc[i][j] = v8f_zero();
  const int ra = tid >> 1, ca = (tid & 1) * 16;
  const int rb = tid >> 2, cb = (tid & 3) * 8;
  for (int k0 = 0; k0 < Kd; k0 += 32) {
    {
      const bf16* src = A + (size_t)(m0 + ra) * Kd + k0 + ca;
      async_copy_b128(&As[ra * 40 + ca], src);
      async_copy_b128(&As[ra * 40 + ca + 8], src + 8);
    }
    {
      const float* src = W + (size_t)(n0 + rb) * Kd + k0 + cb;
      *(bf16x8*)(&Bs[rb * 40 + cb]) = cvt8(src);
      if (k0 + 32 < Kd) __builtin_prefetch((const void*)(src + 32), 0, 1);
    }
    wait_async();
    __syncthreads();
    v16bf af[2], bfv[2];
#pragma unroll
    for (int rt = 0; rt < 2; ++rt) af[rt] = load_a_frag(As, waveM * 32 + rt * 16, 40, 0);
#pragma unroll
    for (int ct = 0; ct < 2; ++ct) bfv[ct] = load_b_frag(Bs, waveN * 32 + ct * 16, 40, 0);
#pragma unroll
    for (int rt = 0; rt < 2; ++rt)
#pragma unroll
      for (int ct = 0; ct < 2; ++ct) acc[rt][ct] = wmma_bf16(af[rt], bfv[ct], acc[rt][ct]);
    __syncthreads();
  }
#pragma unroll
  for (int rt = 0; rt < 2; ++rt)
#pragma unroll
    for (int ct = 0; ct < 2; ++ct)
#pragma unroll
      for (int i = 0; i < 8; ++i) {
        const int row = m0 + waveM * 32 + rt * 16 + i + 8 * hlf;
        const int col = n0 + waveN * 32 + ct * 16 + nn;
        float v = acc[rt][ct][i];
        if (res) v += res[(size_t)row * N + col];
        C[(size_t)row * N + col] = v;
      }
}

// ---------------- flash attention (causal), 32 q-rows per block ----------------

__global__ __launch_bounds__(256) void flash_attn_k(
    const bf16* __restrict__ Q, const bf16* __restrict__ Kmat,
    const bf16* __restrict__ Vt, bf16* __restrict__ O) {
  const int bh = blockIdx.x;
  const int b = bh / NH_, head = bh % NH_;
  const int q0 = blockIdx.y * 32;
  const int tid = threadIdx.x;
  const int w = tid >> 5, lane = tid & 31, hlf = lane >> 4, nn = lane & 15;
  __shared__ bf16  qs[32 * 136];
  __shared__ float sS[32 * 65];
  __shared__ bf16  pbuf[32 * 80];
  __shared__ float alpha_s[32];
  __shared__ float l_s[32];
  {
    const int r = tid >> 3, cb = (tid & 7) * 16;
    const bf16* src = Q + ((size_t)bh * S_ + q0 + r) * HD_ + cb;
    async_copy_b128(&qs[r * 136 + cb], src);
    async_copy_b128(&qs[r * 136 + cb + 8], src + 8);
  }
  float m_i = -1e30f, l_i = 0.0f;
  v8f o[2];
  o[0] = v8f_zero();
  o[1] = v8f_zero();
  const float scale = 0.088388347648318447f;  // 1/sqrt(128)
  const int nch = (q0 + 32 + 63) >> 6;
  wait_async();
  __syncthreads();
  for (int kc = 0; kc < nch; ++kc) {
    const int k0 = kc * 64;
    {  // scores: wave w computes 16x16 tile (rt = w&1, ct4 = w>>1) of the 32x64 chunk
      const int rt = w & 1, ct4 = w >> 1;
      v8f sacc = v8f_zero();
#pragma unroll
      for (int kk = 0; kk < 4; ++kk) {
        v16bf a = load_a_frag(qs, rt * 16, 136, kk * 32);
        const bf16* bp =
            Kmat + ((size_t)bh * S_ + k0 + ct4 * 16 + nn) * HD_ + kk * 32 + hlf * 16;
        v16bf bb = combine16(*(const bf16x8*)bp, *(const bf16x8*)(bp + 8));
        sacc = wmma_bf16(a, bb, sacc);
      }
#pragma unroll
      for (int i = 0; i < 8; ++i) {
        const int row = rt * 16 + i + 8 * hlf;
        const int key = k0 + ct4 * 16 + nn;
        float v = sacc[i] * scale;
        if (key > q0 + row) v = -1e30f;  // causal mask
        sS[row * 65 + ct4 * 16 + nn] = v;
      }
    }
    __syncthreads();
    if (tid < 32) {  // streaming softmax update for one row
      const int row = tid;
      float mx = m_i;
      for (int j = 0; j < 64; ++j) mx = fmaxf(mx, sS[row * 65 + j]);
      const float alpha = __expf(m_i - mx);
      float sum = 0.0f;
      for (int j = 0; j < 64; ++j) {
        const float pv = __expf(sS[row * 65 + j] - mx);
        pbuf[row * 80 + j] = (bf16)pv;
        sum += pv;
      }
      l_i = l_i * alpha + sum;
      m_i = mx;
      alpha_s[row] = alpha;
    }
    __syncthreads();
    // O update: wave w owns output columns w*16..w*16+15
#pragma unroll
    for (int rt = 0; rt < 2; ++rt) {
#pragma unroll
      for (int i = 0; i < 8; ++i) o[rt][i] *= alpha_s[rt * 16 + i + 8 * hlf];
#pragma unroll
      for (int kk = 0; kk < 2; ++kk) {
        v16bf a = load_a_frag(pbuf, rt * 16, 80, kk * 32);
        const bf16* bp =
            Vt + ((size_t)bh * HD_ + w * 16 + nn) * S_ + k0 + kk * 32 + hlf * 16;
        v16bf bb = combine16(*(const bf16x8*)bp, *(const bf16x8*)(bp + 8));
        o[rt] = wmma_bf16(a, bb, o[rt]);
      }
    }
    __syncthreads();
  }
  if (tid < 32) l_s[tid] = l_i;
  __syncthreads();
#pragma unroll
  for (int rt = 0; rt < 2; ++rt)
#pragma unroll
    for (int i = 0; i < 8; ++i) {
      const int row = rt * 16 + i + 8 * hlf;
      const float inv = 1.0f / l_s[row];
      O[((size_t)b * S_ + q0 + row) * H_ + head * HD_ + w * 16 + nn] =
          (bf16)(o[rt][i] * inv);
    }
}

// ---------------- MoE routing ----------------

__global__ void router_k(const bf16* __restrict__ H2, const float* __restrict__ Gw,
                         int* __restrict__ topi, float* __restrict__ topw,
                         int* __restrict__ cnt) {
  const int t = blockIdx.x;
  const int tid = threadIdx.x;
  const int e = tid >> 4, l16 = tid & 15;
  __shared__ float red[256];
  __shared__ float logits[E_];
  float s = 0.0f;
  for (int k = l16; k < H_; k += 16) s += (float)H2[(size_t)t * H_ + k] * Gw[e * H_ + k];
  red[tid] = s;
  __syncthreads();
  if (l16 == 0) {
    float a = 0.0f;
    for (int i = 0; i < 16; ++i) a += red[e * 16 + i];
    logits[e] = a;
  }
  __syncthreads();
  if (tid == 0) {
    float mx = logits[0];
    for (int i = 1; i < E_; ++i) mx = fmaxf(mx, logits[i]);
    float ex[E_], sum = 0.0f;
    for (int i = 0; i < E_; ++i) { ex[i] = __expf(logits[i] - mx); sum += ex[i]; }
    bool used[E_];
    for (int i = 0; i < E_; ++i) used[i] = false;
    for (int j = 0; j < TK_; ++j) {
      int bi = 0;
      float bv = -1.0f;
      for (int i = 0; i < E_; ++i)
        if (!used[i] && ex[i] > bv) { bv = ex[i]; bi = i; }
      used[bi] = true;
      topi[t * TK_ + j] = bi;
      topw[t * TK_ + j] = bv / sum;  // norm_topk_prob = False
      atomicAdd(&cnt[bi], 1);
    }
  }
}

__global__ void scan_k(const int* __restrict__ cnt, int* __restrict__ offs,
                       int* __restrict__ cursor) {
  if (blockIdx.x == 0 && threadIdx.x == 0) {
    int a = 0;
    for (int e = 0; e < E_; ++e) { offs[e] = a; cursor[e] = a; a += cnt[e]; }
  }
}

__global__ void scatter_k(const int* __restrict__ topi, const float* __restrict__ topw,
                          int* __restrict__ cursor, int* __restrict__ tokl,
                          float* __restrict__ wgtl) {
  const int t = blockIdx.x * 256 + threadIdx.x;
  if (t >= T_) return;
  for (int j = 0; j < TK_; ++j) {
    const int e = topi[t * TK_ + j];
    const int pos = atomicAdd(&cursor[e], 1);
    tokl[pos] = t;
    wgtl[pos] = topw[t * TK_ + j];
  }
}

__global__ void seg_k(const bf16* __restrict__ H2, const float* __restrict__ Sw,
                      float* __restrict__ Sig) {
  const int t = blockIdx.x;
  const int tid = threadIdx.x;
  __shared__ float red[256];
  float s = 0.0f;
  for (int k = tid; k < H_; k += 256) s += (float)H2[(size_t)t * H_ + k] * Sw[k];
  red[tid] = s;
  __syncthreads();
  for (int off = 128; off > 0; off >>= 1) {
    if (tid < off) red[tid] += red[tid + off];
    __syncthreads();
  }
  if (tid == 0) Sig[t] = 1.0f / (1.0f + __expf(-red[0]));
}

// ---------------- fused gate/up GEMM (shared expert + gathered experts) ----------------
// Act[slot, Nout] = silu(A@W1^T) * (A@W2^T), A rows gathered via tokl (or identity)
// (register-path A staging: invalid rows must be zero-filled, so no async here)

__global__ __launch_bounds__(256) void gemm_gu(
    const bf16* __restrict__ Hin, const float* __restrict__ W1b,
    const float* __restrict__ W2b, bf16* __restrict__ ActO,
    const int* __restrict__ tokl, const int* __restrict__ offs,
    const int* __restrict__ cnts, int Kd, int Nout, long long wstride) {
  __shared__ bf16 As[128 * 40];
  __shared__ bf16 B1[64 * 40];
  __shared__ bf16 B2[64 * 40];
  const int e = blockIdx.z;
  const int cnt = cnts ? cnts[e] : T_;
  const int m0 = blockIdx.x * 128;
  if (m0 >= cnt) return;
  const int grp = offs ? offs[e] : 0;
  const float* W1 = W1b + (size_t)e * wstride;
  const float* W2 = W2b + (size_t)e * wstride;
  const int n0 = blockIdx.y * 64;
  const int tid = threadIdx.x;
  const int w = tid >> 5, lane = tid & 31, hlf = lane >> 4, nn = lane & 15;
  const int waveM = w & 3, waveN = w >> 2;
  v8f accG[2][2], accU[2][2];
#pragma unroll
  for (int i = 0; i < 2; ++i)
#pragma unroll
    for (int j = 0; j < 2; ++j) { accG[i][j] = v8f_zero(); accU[i][j] = v8f_zero(); }
  const int ra = tid >> 1, ca = (tid & 1) * 16;
  const int rb = tid >> 2, cb = (tid & 3) * 8;
  const bool avalid = (m0 + ra) < cnt;
  int atok = 0;
  if (avalid) atok = tokl ? tokl[grp + m0 + ra] : (m0 + ra);
  for (int k0 = 0; k0 < Kd; k0 += 32) {
    {
      bf16x8 v0 = bf8_zero(), v1 = bf8_zero();
      if (avalid) {
        const bf16* src = Hin + (size_t)atok * Kd + k0 + ca;
        v0 = *(const bf16x8*)(src);
        v1 = *(const bf16x8*)(src + 8);
      }
      *(bf16x8*)(&As[ra * 40 + ca])     = v0;
      *(bf16x8*)(&As[ra * 40 + ca + 8]) = v1;
    }
    {
      const float* s1 = W1 + (size_t)(n0 + rb) * Kd + k0 + cb;
      const float* s2 = W2 + (size_t)(n0 + rb) * Kd + k0 + cb;
      *(bf16x8*)(&B1[rb * 40 + cb]) = cvt8(s1);
      *(bf16x8*)(&B2[rb * 40 + cb]) = cvt8(s2);
      if (k0 + 32 < Kd) {
        __builtin_prefetch((const void*)(s1 + 32), 0, 1);
        __builtin_prefetch((const void*)(s2 + 32), 0, 1);
      }
    }
    __syncthreads();
    v16bf af[2], b1f[2], b2f[2];
#pragma unroll
    for (int rt = 0; rt < 2; ++rt) af[rt] = load_a_frag(As, waveM * 32 + rt * 16, 40, 0);
#pragma unroll
    for (int ct = 0; ct < 2; ++ct) {
      b1f[ct] = load_b_frag(B1, waveN * 32 + ct * 16, 40, 0);
      b2f[ct] = load_b_frag(B2, waveN * 32 + ct * 16, 40, 0);
    }
#pragma unroll
    for (int rt = 0; rt < 2; ++rt)
#pragma unroll
      for (int ct = 0; ct < 2; ++ct) {
        accG[rt][ct] = wmma_bf16(af[rt], b1f[ct], accG[rt][ct]);
        accU[rt][ct] = wmma_bf16(af[rt], b2f[ct], accU[rt][ct]);
      }
    __syncthreads();
  }
#pragma unroll
  for (int rt = 0; rt < 2; ++rt)
#pragma unroll
    for (int ct = 0; ct < 2; ++ct)
#pragma unroll
      for (int i = 0; i < 8; ++i) {
        const int lrow = m0 + waveM * 32 + rt * 16 + i + 8 * hlf;
        if (lrow < cnt) {
          const int col = n0 + waveN * 32 + ct * 16 + nn;
          const float g = accG[rt][ct][i];
          const float u = accU[rt][ct][i];
          const float a = g / (1.0f + __expf(-g)) * u;  // silu(g)*u
          ActO[(size_t)(grp + lrow) * Nout + col] = (bf16)a;
        }
      }
}

// ---------------- expert down-proj: atomic accumulate combine-weighted rows ----------------

__global__ __launch_bounds__(256) void gemm_down(
    const bf16* __restrict__ Act, const float* __restrict__ Wb, float* __restrict__ Moe,
    const int* __restrict__ tokl, const float* __restrict__ wgtl,
    const int* __restrict__ offs, const int* __restrict__ cnts, long long wstride) {
  __shared__ bf16 As[128 * 40];
  __shared__ bf16 Bs[64 * 40];
  const int e = blockIdx.z;
  const int cnt = cnts[e];
  const int m0 = blockIdx.x * 128;
  if (m0 >= cnt) return;
  const int grp = offs[e];
  const float* W = Wb + (size_t)e * wstride;
  const int n0 = blockIdx.y * 64;
  const int tid = threadIdx.x;
  const int w = tid >> 5, lane = tid & 31, hlf = lane >> 4, nn = lane & 15;
  const int waveM = w & 3, waveN = w >> 2;
  v8f acc[2][2];
#pragma unroll
  for (int i = 0; i < 2; ++i)
#pragma unroll
    for (int j = 0; j < 2; ++j) acc[i][j] = v8f_zero();
  const int ra = tid >> 1, ca = (tid & 1) * 16;
  const int rb = tid >> 2, cb = (tid & 3) * 8;
  const bool avalid = (m0 + ra) < cnt;
  for (int k0 = 0; k0 < I_; k0 += 32) {
    {
      bf16x8 v0 = bf8_zero(), v1 = bf8_zero();
      if (avalid) {
        const bf16* src = Act + (size_t)(grp + m0 + ra) * I_ + k0 + ca;
        v0 = *(const bf16x8*)(src);
        v1 = *(const bf16x8*)(src + 8);
      }
      *(bf16x8*)(&As[ra * 40 + ca])     = v0;
      *(bf16x8*)(&As[ra * 40 + ca + 8]) = v1;
    }
    {
      const float* src = W + (size_t)(n0 + rb) * I_ + k0 + cb;
      *(bf16x8*)(&Bs[rb * 40 + cb]) = cvt8(src);
      if (k0 + 32 < I_) __builtin_prefetch((const void*)(src + 32), 0, 1);
    }
    __syncthreads();
    v16bf af[2], bfv[2];
#pragma unroll
    for (int rt = 0; rt < 2; ++rt) af[rt] = load_a_frag(As, waveM * 32 + rt * 16, 40, 0);
#pragma unroll
    for (int ct = 0; ct < 2; ++ct) bfv[ct] = load_b_frag(Bs, waveN * 32 + ct * 16, 40, 0);
#pragma unroll
    for (int rt = 0; rt < 2; ++rt)
#pragma unroll
      for (int ct = 0; ct < 2; ++ct) acc[rt][ct] = wmma_bf16(af[rt], bfv[ct], acc[rt][ct]);
    __syncthreads();
  }
#pragma unroll
  for (int rt = 0; rt < 2; ++rt)
#pragma unroll
    for (int ct = 0; ct < 2; ++ct)
#pragma unroll
      for (int i = 0; i < 8; ++i) {
        const int lrow = m0 + waveM * 32 + rt * 16 + i + 8 * hlf;
        if (lrow < cnt) {
          const int tok = tokl[grp + lrow];
          const float wg = wgtl[grp + lrow];
          const int col = n0 + waveN * 32 + ct * 16 + nn;
          unsafeAtomicAdd(&Moe[(size_t)tok * H_ + col], acc[rt][ct][i] * wg);
        }
      }
}

// ---------------- shared-expert down + final residual combine -> d_out ----------------

__global__ __launch_bounds__(256) void gemm_shared_down(
    const bf16* __restrict__ Sm, const float* __restrict__ W,
    const float* __restrict__ X1, const float* __restrict__ Moe,
    const float* __restrict__ Sig, float* __restrict__ Out) {
  __shared__ bf16 As[128 * 40];
  __shared__ bf16 Bs[64 * 40];
  const int tid = threadIdx.x;
  const int w = tid >> 5, lane = tid & 31, hlf = lane >> 4, nn = lane & 15;
  const int waveM = w & 3, waveN = w >> 2;
  const int m0 = blockIdx.x * 128, n0 = blockIdx.y * 64;
  v8f acc[2][2];
#pragma unroll
  for (int i = 0; i < 2; ++i)
#pragma unroll
    for (int j = 0; j < 2; ++j) acc[i][j] = v8f_zero();
  const int ra = tid >> 1, ca = (tid & 1) * 16;
  const int rb = tid >> 2, cb = (tid & 3) * 8;
  for (int k0 = 0; k0 < I2_; k0 += 32) {
    {
      const bf16* src = Sm + (size_t)(m0 + ra) * I2_ + k0 + ca;
      async_copy_b128(&As[ra * 40 + ca], src);
      async_copy_b128(&As[ra * 40 + ca + 8], src + 8);
    }
    {
      const float* src = W + (size_t)(n0 + rb) * I2_ + k0 + cb;
      *(bf16x8*)(&Bs[rb * 40 + cb]) = cvt8(src);
      if (k0 + 32 < I2_) __builtin_prefetch((const void*)(src + 32), 0, 1);
    }
    wait_async();
    __syncthreads();
    v16bf af[2], bfv[2];
#pragma unroll
    for (int rt = 0; rt < 2; ++rt) af[rt] = load_a_frag(As, waveM * 32 + rt * 16, 40, 0);
#pragma unroll
    for (int ct = 0; ct < 2; ++ct) bfv[ct] = load_b_frag(Bs, waveN * 32 + ct * 16, 40, 0);
#pragma unroll
    for (int rt = 0; rt < 2; ++rt)
#pragma unroll
      for (int ct = 0; ct < 2; ++ct) acc[rt][ct] = wmma_bf16(af[rt], bfv[ct], acc[rt][ct]);
    __syncthreads();
  }
#pragma unroll
  for (int rt = 0; rt < 2; ++rt)
#pragma unroll
    for (int ct = 0; ct < 2; ++ct)
#pragma unroll
      for (int i = 0; i < 8; ++i) {
        const int row = m0 + waveM * 32 + rt * 16 + i + 8 * hlf;
        const int col = n0 + waveN * 32 + ct * 16 + nn;
        const size_t off = (size_t)row * H_ + col;
        Out[off] = X1[off] + Moe[off] + acc[rt][ct][i] * Sig[row];
      }
}

// ---------------- host-side orchestration ----------------

extern "C" void kernel_launch(void* const* d_in, const int* in_sizes, int n_in,
                              void* d_out, int out_size, void* d_ws, size_t ws_size,
                              hipStream_t stream) {
  (void)in_sizes; (void)n_in; (void)out_size; (void)ws_size;
  const float* x0   = (const float*)d_in[0];
  const int*   pos  = (const int*)d_in[2];
  const float* wq   = (const float*)d_in[3];
  const float* wk   = (const float*)d_in[4];
  const float* wv   = (const float*)d_in[5];
  const float* wo   = (const float*)d_in[6];
  const float* ln1  = (const float*)d_in[7];
  const float* ln2  = (const float*)d_in[8];
  const float* gw   = (const float*)d_in[9];
  const float* egw  = (const float*)d_in[10];
  const float* euw  = (const float*)d_in[11];
  const float* edw  = (const float*)d_in[12];
  const float* sgw  = (const float*)d_in[13];
  const float* suw  = (const float*)d_in[14];
  const float* sdw  = (const float*)d_in[15];
  const float* segw = (const float*)d_in[16];
  float* out = (float*)d_out;

  char* p = (char*)d_ws;
  auto alloc = [&](size_t bytes) -> void* {
    void* r = (void*)p;
    p += (bytes + 255) & ~(size_t)255;
    return r;
  };
  bf16*  h1   = (bf16*)alloc((size_t)T_ * H_ * 2);
  float* qf   = (float*)alloc((size_t)T_ * H_ * 4);
  float* kf   = (float*)alloc((size_t)T_ * H_ * 4);
  float* vf   = (float*)alloc((size_t)T_ * H_ * 4);
  bf16*  qb   = (bf16*)alloc((size_t)T_ * H_ * 2);
  bf16*  kb   = (bf16*)alloc((size_t)T_ * H_ * 2);
  bf16*  vt   = (bf16*)alloc((size_t)T_ * H_ * 2);
  bf16*  attn = (bf16*)alloc((size_t)T_ * H_ * 2);
  float* x1   = (float*)alloc((size_t)T_ * H_ * 4);
  bf16*  h2   = (bf16*)alloc((size_t)T_ * H_ * 2);
  int*   topi = (int*)alloc((size_t)T_ * TK_ * 4);
  float* topw = (float*)alloc((size_t)T_ * TK_ * 4);
  int*   cnt  = (int*)alloc(E_ * 4);
  int*   offs = (int*)alloc(E_ * 4);
  int*   cur  = (int*)alloc(E_ * 4);
  int*   tokl = (int*)alloc((size_t)SLOTS_ * 4);
  float* wgtl = (float*)alloc((size_t)SLOTS_ * 4);
  bf16*  act  = (bf16*)alloc((size_t)SLOTS_ * I_ * 2);
  float* moe  = (float*)alloc((size_t)T_ * H_ * 4);
  bf16*  sm   = (bf16*)alloc((size_t)T_ * I2_ * 2);
  float* sig  = (float*)alloc((size_t)T_ * 4);

  const dim3 gqkv(T_ / 128, H_ / 64);

  zero_k<<<(T_ * H_ + 255) / 256, 256, 0, stream>>>(moe, cnt);
  rmsnorm_k<<<T_, 256, 0, stream>>>(x0, ln1, h1);
  gemm_f32<<<gqkv, 256, 0, stream>>>(h1, wq, qf, nullptr, H_, H_);
  gemm_f32<<<gqkv, 256, 0, stream>>>(h1, wk, kf, nullptr, H_, H_);
  gemm_f32<<<gqkv, 256, 0, stream>>>(h1, wv, vf, nullptr, H_, H_);
  rope_k<<<(B_ * S_ * NH_ * 64) / 256, 256, 0, stream>>>(qf, kf, pos, qb, kb);
  vtr_k<<<(B_ * NH_ * HD_ * S_) / 256, 256, 0, stream>>>(vf, vt);
  flash_attn_k<<<dim3(B_ * NH_, S_ / 32), 256, 0, stream>>>(qb, kb, vt, attn);
  gemm_f32<<<gqkv, 256, 0, stream>>>(attn, wo, x1, x0, H_, H_);
  rmsnorm_k<<<T_, 256, 0, stream>>>(x1, ln2, h2);
  router_k<<<T_, 256, 0, stream>>>(h2, gw, topi, topw, cnt);
  scan_k<<<1, 1, 0, stream>>>(cnt, offs, cur);
  scatter_k<<<(T_ + 255) / 256, 256, 0, stream>>>(topi, topw, cur, tokl, wgtl);
  gemm_gu<<<dim3(T_ / 128, I2_ / 64, 1), 256, 0, stream>>>(
      h2, sgw, suw, sm, nullptr, nullptr, nullptr, H_, I2_, 0LL);
  gemm_gu<<<dim3(T_ / 128, I_ / 64, E_), 256, 0, stream>>>(
      h2, egw, euw, act, tokl, offs, cnt, H_, I_, (long long)I_ * H_);
  seg_k<<<T_, 256, 0, stream>>>(h2, segw, sig);
  gemm_down<<<dim3(T_ / 128, H_ / 64, E_), 256, 0, stream>>>(
      act, edw, moe, tokl, wgtl, offs, cnt, (long long)H_ * I_);
  gemm_shared_down<<<dim3(T_ / 128, H_ / 64), 256, 0, stream>>>(sm, sdw, x1, moe, sig, out);
}
